// Track_89644557402870
// MI455X (gfx1250) — matrix operand order repaired and checked by reading, not compiled, not run on hardware
//
#include <hip/hip_runtime.h>
#include <hip/hip_bf16.h>
#include <math.h>

typedef __attribute__((ext_vector_type(16))) __bf16       v16bf;
typedef __attribute__((ext_vector_type(8)))  float        v8f;
typedef __attribute__((ext_vector_type(8)))  unsigned int v8u;
typedef unsigned short u16;

__device__ __forceinline__ unsigned int pk_bf16(float a, float b) {
    unsigned int ua = __float_as_uint(a);
    unsigned int ub = __float_as_uint(b);
    ua = (ua + 0x7FFFu + ((ua >> 16) & 1u)) >> 16;
    ub = (ub + 0x7FFFu + ((ub >> 16) & 1u)) >> 16;
    return ua | (ub << 16);
}

__device__ __forceinline__ u16 cvt1_bf16(float a) {
    unsigned int ua = __float_as_uint(a);
    return (u16)((ua + 0x7FFFu + ((ua >> 16) & 1u)) >> 16);
}

// ---------------------------------------------------------------------------
// One-time f32 -> bf16 conversion (memory bound, 8 elems/thread). n % 8 == 0.
// ---------------------------------------------------------------------------
__global__ void cvt_bf16_k(const float* __restrict__ src, u16* __restrict__ dst, int n)
{
    int i = (blockIdx.x * blockDim.x + threadIdx.x) * 8;
    if (i >= n) return;
    float4 a = *(const float4*)(src + i);
    float4 b = *(const float4*)(src + i + 4);
    uint4 o;
    o.x = pk_bf16(a.x, a.y);
    o.y = pk_bf16(a.z, a.w);
    o.z = pk_bf16(b.x, b.y);
    o.w = pk_bf16(b.z, b.w);
    *(uint4*)(dst + i) = o;
}

// ---------------------------------------------------------------------------
// C[M x N] = (Cin ? Cin : 0) + A[M x K] * B[N x K]^T        (A, B bf16; C f32)
// Workgroup = 4 waves = 64x128 macro-tile; each wave owns a 16x128 slice.
// Per K=32 step the 128x32 B chunk is staged in LDS once (8 KB) and consumed
// by all 4 waves. Staging is register-double-buffered against the global
// loads of the NEXT chunk, and the per-tile B fragments use a 2-deep
// ping-pong register pipeline so each WMMA overlaps the next fragment's
// ds_load latency (wait dscnt<=2 instead of 0).
// Requires M%64==0, N%128==0, K%32==0, lda/ldb elements multiples of 8.
// ---------------------------------------------------------------------------
__global__ void __launch_bounds__(128)
wmma_gemm_k(const u16* __restrict__ A, int lda,
            const u16* __restrict__ B, int ldb,
            const float* __restrict__ Cin, int ldcin,
            float* __restrict__ Cout, int ldcout,
            int K)
{
    __shared__ u16 lb[128 * 32];            // 128 cols x 32 K, 64 B per col

    const int tid  = threadIdx.x;           // 0..127
    const int wid  = tid >> 5;              // wave 0..3 -> M sub-tile
    const int lane = tid & 31;
    const int half = lane >> 4;             // K-half selector
    const int l16  = lane & 15;
    const int n0   = blockIdx.x * 128;
    const int m0   = blockIdx.y * 64 + wid * 16;
    const int crow = m0 + half * 8;

    v8f acc[8];
    if (Cin) {
        #pragma unroll
        for (int t = 0; t < 8; ++t)
            #pragma unroll
            for (int r = 0; r < 8; ++r)
                acc[t][r] = Cin[(size_t)(crow + r) * ldcin + n0 + t * 16 + l16];
    } else {
        #pragma unroll
        for (int t = 0; t < 8; ++t)
            #pragma unroll
            for (int r = 0; r < 8; ++r)
                acc[t][r] = 0.0f;
    }

    const u16* arow = A + (size_t)(m0 + l16) * lda + half * 16;
    const u16* bcol = B + (size_t)(n0 + tid) * ldb;   // this thread's staging col
    u16*       lcol = lb + tid * 32;
    const u16* lfrg = lb + l16 * 32 + half * 16;      // fragment base (tile 0)

    // prologue: preload first staging chunk into registers
    uint4 c0 = *(const uint4*)(bcol);
    uint4 c1 = *(const uint4*)(bcol + 8);
    uint4 c2 = *(const uint4*)(bcol + 16);
    uint4 c3 = *(const uint4*)(bcol + 24);

    for (int k0 = 0; k0 < K; k0 += 32) {
        __syncthreads();                    // prev iteration's LDS reads done
        *(uint4*)(lcol)      = c0;
        *(uint4*)(lcol + 8)  = c1;
        *(uint4*)(lcol + 16) = c2;
        *(uint4*)(lcol + 24) = c3;
        // register double-buffer: issue next chunk's global loads now
        if (k0 + 32 < K) {
            const u16* gp = bcol + k0 + 32;
            c0 = *(const uint4*)(gp);
            c1 = *(const uint4*)(gp + 8);
            c2 = *(const uint4*)(gp + 16);
            c3 = *(const uint4*)(gp + 24);
        }
        // A fragment: two b128 global loads (overlap the barrier)
        uint4 a0 = *(const uint4*)(arow + k0);
        uint4 a1 = *(const uint4*)(arow + k0 + 8);
        __syncthreads();                    // staging visible to all waves

        v8u au;
        au[0] = a0.x; au[1] = a0.y; au[2] = a0.z; au[3] = a0.w;
        au[4] = a1.x; au[5] = a1.y; au[6] = a1.z; au[7] = a1.w;
        v16bf afrag = __builtin_bit_cast(v16bf, au);

        // 2-deep ping-pong over the 8 N-tiles: prefetch frag t+1 before wmma t
        uint4 ba[2], bb[2];
        ba[0] = *(const uint4*)(lfrg);
        bb[0] = *(const uint4*)(lfrg + 8);
        #pragma unroll
        for (int t = 0; t < 8; ++t) {
            if (t < 7) {
                const u16* lp = lfrg + (t + 1) * 512;     // next tile (16*32 u16)
                ba[(t + 1) & 1] = *(const uint4*)(lp);
                bb[(t + 1) & 1] = *(const uint4*)(lp + 8);
            }
            v8u bu;
            bu[0] = ba[t & 1].x; bu[1] = ba[t & 1].y;
            bu[2] = ba[t & 1].z; bu[3] = ba[t & 1].w;
            bu[4] = bb[t & 1].x; bu[5] = bb[t & 1].y;
            bu[6] = bb[t & 1].z; bu[7] = bb[t & 1].w;
            v16bf bfrag = __builtin_bit_cast(v16bf, bu);
            acc[t] = __builtin_amdgcn_wmma_f32_16x16x32_bf16(
                false, afrag, false, bfrag, (short)0, acc[t], false, false);
        }
    }

    #pragma unroll
    for (int t = 0; t < 8; ++t)
        #pragma unroll
        for (int r = 0; r < 8; ++r)
            Cout[(size_t)(crow + r) * ldcout + n0 + t * 16 + l16] = acc[t][r];
}

// ---------------------------------------------------------------------------
// Tiny input projection for din=3: G[r,n] = sum_{k<3} X[r,k] * W[n,k]
// ---------------------------------------------------------------------------
__global__ void small_xw_k(const float* __restrict__ X,   // R x 3
                           const float* __restrict__ W,   // 1024 x 3
                           float* __restrict__ G, int R)
{
    int idx = blockIdx.x * blockDim.x + threadIdx.x;
    if (idx >= R * 1024) return;
    int r = idx >> 10, n = idx & 1023;
    const float* x = X + (size_t)r * 3;
    const float* w = W + (size_t)n * 3;
    G[idx] = x[0] * w[0] + x[1] * w[1] + x[2] * w[2];
}

__device__ __forceinline__ float sigmoidf(float x) {
    return 1.0f / (1.0f + __expf(-x));
}

// ---------------------------------------------------------------------------
// LSTM cell epilogue: gates (B x 1024, i|f|g|o) + biases -> c', h' (f32 + bf16)
// ---------------------------------------------------------------------------
__global__ void lstm_cell_k(const float* __restrict__ gates,
                            const float* __restrict__ bih,
                            const float* __restrict__ bhh,
                            const float* __restrict__ c_in,
                            float* __restrict__ c_out,
                            float* __restrict__ h_out,
                            u16* __restrict__ h_bf, int B)
{
    int idx = blockIdx.x * blockDim.x + threadIdx.x;
    if (idx >= B * 256) return;
    int b = idx >> 8, n = idx & 255;
    const float* g = gates + (size_t)b * 1024;
    float gi = g[n]       + bih[n]       + bhh[n];
    float gf = g[n + 256] + bih[n + 256] + bhh[n + 256];
    float gg = g[n + 512] + bih[n + 512] + bhh[n + 512];
    float go = g[n + 768] + bih[n + 768] + bhh[n + 768];
    float c  = sigmoidf(gf) * c_in[idx] + sigmoidf(gi) * tanhf(gg);
    float h  = sigmoidf(go) * tanhf(c);
    c_out[idx] = c;
    h_out[idx] = h;
    h_bf[idx]  = cvt1_bf16(h);
}

// ---------------------------------------------------------------------------
// delta = (fuse2 + f2_b) @ out_W^T + out_b ; pred += delta ; write d_out[b,t,:]
// ---------------------------------------------------------------------------
__global__ void out_proj_k(const float* __restrict__ fuse2,
                           const float* __restrict__ f2b,
                           const float* __restrict__ oW,   // 3 x 256
                           const float* __restrict__ ob,   // 3
                           float* __restrict__ pred,       // B x 3 carry
                           float* __restrict__ out,        // B x T x 3
                           int t, int T, int B)
{
    int b = blockIdx.x * blockDim.x + threadIdx.x;
    if (b >= B) return;
    const float* f = fuse2 + (size_t)b * 256;
    float d0 = ob[0], d1 = ob[1], d2 = ob[2];
    for (int n = 0; n < 256; ++n) {
        float v = f[n] + f2b[n];
        d0 += v * oW[n];
        d1 += v * oW[256 + n];
        d2 += v * oW[512 + n];
    }
    float p0 = pred[b * 3 + 0] + d0;
    float p1 = pred[b * 3 + 1] + d1;
    float p2 = pred[b * 3 + 2] + d2;
    pred[b * 3 + 0] = p0; pred[b * 3 + 1] = p1; pred[b * 3 + 2] = p2;
    float* o = out + (size_t)(b * T + t) * 3;
    o[0] = p0; o[1] = p1; o[2] = p2;
}

__global__ void fill0_k(float* __restrict__ p, int n)
{
    int i = blockIdx.x * blockDim.x + threadIdx.x;
    if (i < n) p[i] = 0.0f;
}

// ---------------------------------------------------------------------------
extern "C" void kernel_launch(void* const* d_in, const int* in_sizes, int n_in,
                              void* d_out, int out_size, void* d_ws, size_t ws_size,
                              hipStream_t stream)
{
    (void)in_sizes; (void)n_in; (void)out_size; (void)ws_size;

    const float* enc_pos = (const float*)d_in[0];   // (512,16,3)
    const float* enc_sal = (const float*)d_in[1];   // (512,16,4096)
    const float* dec_pos = (const float*)d_in[2];   // (512,1,3)
    const float* dec_sal = (const float*)d_in[3];   // (512,25,4096)
    const float* pe_Wih = (const float*)d_in[4],  *pe_Whh = (const float*)d_in[5],
               * pe_bih = (const float*)d_in[6],  *pe_bhh = (const float*)d_in[7];
    const float* se_Wih = (const float*)d_in[8],  *se_Whh = (const float*)d_in[9],
               * se_bih = (const float*)d_in[10], *se_bhh = (const float*)d_in[11];
    const float* fe_Wih = (const float*)d_in[12], *fe_Whh = (const float*)d_in[13],
               * fe_bih = (const float*)d_in[14], *fe_bhh = (const float*)d_in[15];
    const float* pd_Wih = (const float*)d_in[16], *pd_Whh = (const float*)d_in[17],
               * pd_bih = (const float*)d_in[18], *pd_bhh = (const float*)d_in[19];
    const float* sd_Wih = (const float*)d_in[20], *sd_Whh = (const float*)d_in[21],
               * sd_bih = (const float*)d_in[22], *sd_bhh = (const float*)d_in[23];
    const float* fd_Wih = (const float*)d_in[24], *fd_Whh = (const float*)d_in[25],
               * fd_bih = (const float*)d_in[26], *fd_bhh = (const float*)d_in[27];
    const float* f2_W = (const float*)d_in[28], *f2_b = (const float*)d_in[29];
    const float* out_W = (const float*)d_in[30], *out_b = (const float*)d_in[31];

    const int B = 512, M = 16, H = 256, G4 = 1024, SAL = 4096, T = 25;

    // ---- workspace carve-up (256B aligned slices) ----
    char* wsb = (char*)d_ws;
    size_t off = 0;
    auto carve = [&](size_t bytes) {
        void* p = wsb + off;
        off += (bytes + 255) & ~(size_t)255;
        return p;
    };
    // f32 scratch
    float* G_pe  = (float*)carve((size_t)B * M * G4 * 4);
    float* G_se  = (float*)carve((size_t)B * M * G4 * 4);
    float* G_sd  = (float*)carve((size_t)B * T * G4 * 4);
    float* G_fd  = (float*)carve((size_t)B * G4 * 4);
    float* gat_a = (float*)carve((size_t)B * G4 * 4);
    float* gat_b = (float*)carve((size_t)B * G4 * 4);
    float* gat_c = (float*)carve((size_t)B * G4 * 4);
    float* st    = (float*)carve((size_t)8 * B * H * 4);  // 8 contiguous state bufs
    float* h_pe = st + 0 * B * H, *c_pe = st + 1 * B * H;
    float* h_se = st + 2 * B * H, *c_se = st + 3 * B * H;
    float* h_fe = st + 4 * B * H, *c_fe = st + 5 * B * H;
    float* h_fd = st + 6 * B * H, *c_dm = st + 7 * B * H;
    float* fuse2 = (float*)carve((size_t)B * H * 4);
    float* pred  = (float*)carve((size_t)B * 4 * 4);
    // bf16 operands
    u16* enc_sal_bf = (u16*)carve((size_t)B * M * SAL * 2);
    u16* dec_sal_bf = (u16*)carve((size_t)B * T * SAL * 2);
    u16* se_Wih_bf  = (u16*)carve((size_t)G4 * SAL * 2);
    u16* sd_Wih_bf  = (u16*)carve((size_t)G4 * SAL * 2);
    u16* fe_Wih_bf  = (u16*)carve((size_t)G4 * 2 * H * 2);
    u16* fd_Wih_bf  = (u16*)carve((size_t)G4 * 2 * H * 2);
    u16* pe_Whh_bf  = (u16*)carve((size_t)G4 * H * 2);
    u16* se_Whh_bf  = (u16*)carve((size_t)G4 * H * 2);
    u16* fe_Whh_bf  = (u16*)carve((size_t)G4 * H * 2);
    u16* pd_Whh_bf  = (u16*)carve((size_t)G4 * H * 2);
    u16* sd_Whh_bf  = (u16*)carve((size_t)G4 * H * 2);
    u16* fd_Whh_bf  = (u16*)carve((size_t)G4 * H * 2);
    u16* f2_W_bf    = (u16*)carve((size_t)H * H * 2);
    u16* hbf        = (u16*)carve((size_t)4 * B * H * 2); // 4 contiguous bf16 h bufs
    u16* h_pe_bf = hbf + 0 * B * H, *h_se_bf = hbf + 1 * B * H;
    u16* h_fe_bf = hbf + 2 * B * H, *h_fd_bf = hbf + 3 * B * H;

    const dim3 blk(128);                         // 4 waves -> 64x128 macro-tile
    const dim3 gRec(G4 / 128, B / 64);
    const int  cellG = (B * H) / 256;

    auto cvt = [&](const float* s, u16* d, int n) {
        cvt_bf16_k<<<(n / 8 + 255) / 256, 256, 0, stream>>>(s, d, n);
    };

    // zero f32 states + bf16 h buffers (hbf is 4*B*H u16 = 2*B*H f32 words)
    fill0_k<<<(8 * B * H) / 256, 256, 0, stream>>>(st, 8 * B * H);
    fill0_k<<<(2 * B * H) / 256, 256, 0, stream>>>((float*)hbf, 2 * B * H);
    // pred carry <- decoder_position_inputs[:, 0]
    hipMemcpyAsync(pred, dec_pos, (size_t)B * 3 * sizeof(float),
                   hipMemcpyDeviceToDevice, stream);

    // ---- one-time bf16 conversions ----
    cvt(enc_sal, enc_sal_bf, B * M * SAL);
    cvt(dec_sal, dec_sal_bf, B * T * SAL);
    cvt(se_Wih, se_Wih_bf, G4 * SAL);
    cvt(sd_Wih, sd_Wih_bf, G4 * SAL);
    cvt(fe_Wih, fe_Wih_bf, G4 * 2 * H);
    cvt(fd_Wih, fd_Wih_bf, G4 * 2 * H);
    cvt(pe_Whh, pe_Whh_bf, G4 * H);
    cvt(se_Whh, se_Whh_bf, G4 * H);
    cvt(fe_Whh, fe_Whh_bf, G4 * H);
    cvt(pd_Whh, pd_Whh_bf, G4 * H);
    cvt(sd_Whh, sd_Whh_bf, G4 * H);
    cvt(fd_Whh, fd_Whh_bf, G4 * H);
    cvt(f2_W, f2_W_bf, H * H);

    // ---- hoisted input projections (big K=4096 WMMA GEMMs) ----
    small_xw_k<<<(B * M * G4) / 256, 256, 0, stream>>>(enc_pos, pe_Wih, G_pe, B * M);
    wmma_gemm_k<<<dim3(G4 / 128, (B * M) / 64), blk, 0, stream>>>(
        enc_sal_bf, SAL, se_Wih_bf, SAL, nullptr, 0, G_se, G4, SAL);
    wmma_gemm_k<<<dim3(G4 / 128, (B * T) / 64), blk, 0, stream>>>(
        dec_sal_bf, SAL, sd_Wih_bf, SAL, nullptr, 0, G_sd, G4, SAL);

    // ---- encoder: 16 steps of pe / se / fe ----
    for (int t = 0; t < M; ++t) {
        wmma_gemm_k<<<gRec, blk, 0, stream>>>(h_pe_bf, H, pe_Whh_bf, H,
            G_pe + t * G4, M * G4, gat_a, G4, H);
        lstm_cell_k<<<cellG, 256, 0, stream>>>(gat_a, pe_bih, pe_bhh,
            c_pe, c_pe, h_pe, h_pe_bf, B);

        wmma_gemm_k<<<gRec, blk, 0, stream>>>(h_se_bf, H, se_Whh_bf, H,
            G_se + t * G4, M * G4, gat_b, G4, H);
        lstm_cell_k<<<cellG, 256, 0, stream>>>(gat_b, se_bih, se_bhh,
            c_se, c_se, h_se, h_se_bf, B);

        // fe input = concat(h_se, h_pe): split K=512 Wih into two K=256 gemms
        wmma_gemm_k<<<gRec, blk, 0, stream>>>(h_se_bf, H, fe_Wih_bf, 2 * H,
            nullptr, 0, gat_c, G4, H);
        wmma_gemm_k<<<gRec, blk, 0, stream>>>(h_pe_bf, H, fe_Wih_bf + H, 2 * H,
            gat_c, G4, gat_c, G4, H);
        wmma_gemm_k<<<gRec, blk, 0, stream>>>(h_fe_bf, H, fe_Whh_bf, H,
            gat_c, G4, gat_c, G4, H);
        lstm_cell_k<<<cellG, 256, 0, stream>>>(gat_c, fe_bih, fe_bhh,
            c_fe, c_fe, h_fe, h_fe_bf, B);
    }

    // fd uses the fixed encoder state hf every step: hf @ fd_Whh^T once
    wmma_gemm_k<<<gRec, blk, 0, stream>>>(h_fe_bf, H, fd_Whh_bf, H,
        nullptr, 0, G_fd, G4, H);

    // ---- decoder: 25 steps ----
    float* dout = (float*)d_out;
    for (int t = 0; t < T; ++t) {
        small_xw_k<<<(B * G4) / 256, 256, 0, stream>>>(pred, pd_Wih, gat_a, B);
        wmma_gemm_k<<<gRec, blk, 0, stream>>>(h_pe_bf, H, pd_Whh_bf, H,
            gat_a, G4, gat_a, G4, H);
        lstm_cell_k<<<cellG, 256, 0, stream>>>(gat_a, pd_bih, pd_bhh,
            c_pe, c_pe, h_pe, h_pe_bf, B);

        wmma_gemm_k<<<gRec, blk, 0, stream>>>(h_se_bf, H, sd_Whh_bf, H,
            G_sd + t * G4, T * G4, gat_b, G4, H);
        lstm_cell_k<<<cellG, 256, 0, stream>>>(gat_b, sd_bih, sd_bhh,
            c_se, c_se, h_se, h_se_bf, B);

        // fd input = concat(h_se(new), h_pe(new)); base = constant hf @ fd_Whh^T
        wmma_gemm_k<<<gRec, blk, 0, stream>>>(h_se_bf, H, fd_Wih_bf, 2 * H,
            G_fd, G4, gat_c, G4, H);
        wmma_gemm_k<<<gRec, blk, 0, stream>>>(h_pe_bf, H, fd_Wih_bf + H, 2 * H,
            gat_c, G4, gat_c, G4, H);
        lstm_cell_k<<<cellG, 256, 0, stream>>>(gat_c, fd_bih, fd_bhh,
            c_fe, c_dm, h_fd, h_fd_bf, B);

        wmma_gemm_k<<<dim3(H / 128, B / 64), blk, 0, stream>>>(h_fd_bf, H,
            f2_W_bf, H, nullptr, 0, fuse2, H, H);
        out_proj_k<<<B / 64, 64, 0, stream>>>(fuse2, f2_b, out_W, out_b,
            pred, dout, t, T, B);
    }
}
